// SelfAttention_25013889532556
// MI455X (gfx1250) — matrix operand order repaired and checked by reading, not compiled, not run on hardware
//
#include <hip/hip_runtime.h>

#define S_LEN   2048
#define D_DIM   128
#define BH_DIM  32
#define Q_TILE  64          // query rows per block
#define N_WAVES 4
#define KT      32          // keys per tile
#define K_STRIDE  136       // bf16 elems per Klds row (128 + 8 pad: 16B align kept, bank conflicts killed)
#define VT_STRIDE 40        // bf16 elems per Vt row  (32 keys + 8 pad)
#define P_STRIDE  40        // bf16 elems per P row   (32 + 8 pad)

typedef __bf16 v16bf  __attribute__((ext_vector_type(16)));
typedef __bf16 bf16x8 __attribute__((ext_vector_type(8)));
typedef __bf16 bf16x4 __attribute__((ext_vector_type(4)));
typedef float  v8f    __attribute__((ext_vector_type(8)));
typedef float  f32x4  __attribute__((ext_vector_type(4)));

__global__ __launch_bounds__(128)
void fa_fwd_bf16wmma(const float* __restrict__ Q, const float* __restrict__ K,
                     const float* __restrict__ V, float* __restrict__ O) {
    // double-buffered bf16 tiles for software pipelining
    __shared__ __align__(16) __bf16 Klds[2][KT * K_STRIDE];         // K tile, row-major [key][d]
    __shared__ __align__(16) __bf16 Vt[2][D_DIM * VT_STRIDE];       // V tile, transposed [d][key]
    __shared__ __align__(16) __bf16 Pw[N_WAVES * 16 * P_STRIDE];    // per-wave P scratch [row][key]

    const int tid  = threadIdx.x;
    const int wave = tid >> 5;
    const int lane = tid & 31;
    const int half = lane >> 4;   // 0 / 1 (lanes 0-15 vs 16-31)
    const int l16  = lane & 15;

    const int bh = blockIdx.y;
    const int qb = blockIdx.x * Q_TILE;
    const int qw = qb + wave * 16;        // wave's first query row

    const float scale = 0.08838834764831845f;  // 1/sqrt(128)

    const size_t base = (size_t)bh * S_LEN * D_DIM;
    const float* Qb = Q + base;
    const float* Kb = K + base;
    const float* Vb = V + base;
    float*       Ob = O + base;

    // staging geometry: 128 threads x 8 float4 = 32x128 fp32 tile
    const int stg_f   = tid * 4;               // flat float idx of j=0 element
    // per-j: f = (tid + j*128)*4 -> key = f>>7, d = f&127

    // ---- Q tile into registers in 16-bit A-matrix layout (16 x 128 as 4 K-chunks of 32) ----
    // lane holds row M = l16; lanes<16: K = {c*32+0..7, c*32+16..23}; lanes>=16: +8
    v16bf qa[4];
    {
        const float* qrow = Qb + (size_t)(qw + l16) * D_DIM;
        const int kb = half * 8;
        #pragma unroll
        for (int c = 0; c < 4; ++c) {
            const float* p0 = qrow + c * 32 + kb;
            #pragma unroll
            for (int e = 0; e < 8; ++e) qa[c][e] = (__bf16)p0[e];
            const float* p1 = p0 + 16;
            #pragma unroll
            for (int e = 0; e < 8; ++e) qa[c][8 + e] = (__bf16)p1[e];
        }
    }

    // ---- accumulators: O = 16x128 f32 in 8 C-layout chunks; m/l per row ----
    v8f o_acc[8];
    #pragma unroll
    for (int c = 0; c < 8; ++c)
        #pragma unroll
        for (int r = 0; r < 8; ++r) o_acc[c][r] = 0.0f;
    float m_i[8], l_i[8];
    #pragma unroll
    for (int r = 0; r < 8; ++r) { m_i[r] = -1e30f; l_i[r] = 0.0f; }

    const int ntiles = (qb + Q_TILE) / KT;   // causal: only keys < qb+64 matter (uniform per block)
    __bf16* PwW = &Pw[wave * 16 * P_STRIDE];

    // prefetch registers (held across compute to hide HBM latency)
    f32x4 kpre[8], vpre[8];

    // ---- prologue: load + stage tile 0 into buffer 0 ----
    #pragma unroll
    for (int j = 0; j < 8; ++j) {
        const int f   = stg_f + j * 512;
        const int key = f >> 7;
        const int d   = f & 127;
        kpre[j] = *(const f32x4*)(Kb + (size_t)key * D_DIM + d);
        vpre[j] = *(const f32x4*)(Vb + (size_t)key * D_DIM + d);
    }
    #pragma unroll
    for (int j = 0; j < 8; ++j) {
        const int f   = stg_f + j * 512;
        const int key = f >> 7;
        const int d   = f & 127;
        bf16x4 k4;
        k4[0] = (__bf16)kpre[j][0]; k4[1] = (__bf16)kpre[j][1];
        k4[2] = (__bf16)kpre[j][2]; k4[3] = (__bf16)kpre[j][3];
        *(bf16x4*)(&Klds[0][key * K_STRIDE + d]) = k4;
        Vt[0][(d + 0) * VT_STRIDE + key] = (__bf16)vpre[j][0];
        Vt[0][(d + 1) * VT_STRIDE + key] = (__bf16)vpre[j][1];
        Vt[0][(d + 2) * VT_STRIDE + key] = (__bf16)vpre[j][2];
        Vt[0][(d + 3) * VT_STRIDE + key] = (__bf16)vpre[j][3];
    }
    __syncthreads();

    for (int kt = 0; kt < ntiles; ++kt) {
        const int kbase = kt * KT;
        const int cur   = kt & 1;
        const int nxt   = cur ^ 1;
        const bool have_next = (kt + 1) < ntiles;   // block-uniform

        // ---- issue next tile's global loads early (latency hidden by compute) ----
        if (have_next) {
            const int nbase = kbase + KT;
            #pragma unroll
            for (int j = 0; j < 8; ++j) {
                const int f   = stg_f + j * 512;
                const int key = f >> 7;
                const int d   = f & 127;
                kpre[j] = *(const f32x4*)(Kb + (size_t)(nbase + key) * D_DIM + d);
                vpre[j] = *(const f32x4*)(Vb + (size_t)(nbase + key) * D_DIM + d);
            }
        }

        // ---- compute on buffer `cur` (wave-uniform causal skip -> EXEC stays full) ----
        if (kbase < qw + 16) {
            // preload ALL QK B-operands, then chain WMMAs (one dscnt wait, not eight)
            v16bf kbm[2][4];
            #pragma unroll
            for (int sb = 0; sb < 2; ++sb) {
                const int lkey = sb * 16 + l16;   // B-layout: lane -> column N = key
                #pragma unroll
                for (int c = 0; c < 4; ++c) {
                    const __bf16* kp = &Klds[cur][lkey * K_STRIDE + c * 32 + half * 8];
                    const bf16x8 lo = *(const bf16x8*)(kp);
                    const bf16x8 hi = *(const bf16x8*)(kp + 16);
                    #pragma unroll
                    for (int e = 0; e < 8; ++e) { kbm[sb][c][e] = lo[e]; kbm[sb][c][8 + e] = hi[e]; }
                }
            }
            v8f s[2];
            #pragma unroll
            for (int sb = 0; sb < 2; ++sb) {
                v8f acc;
                #pragma unroll
                for (int r = 0; r < 8; ++r) acc[r] = 0.0f;
                #pragma unroll
                for (int c = 0; c < 4; ++c)
                    acc = __builtin_amdgcn_wmma_f32_16x16x32_bf16(
                              false, qa[c], false, kbm[sb][c], (short)0, acc, false, false);
                s[sb] = acc;
            }

            // ---- scale + causal mask (C-layout: row = r + 8*half, col = lane%16) ----
            const bool need_mask = (kbase + 31) > qw;
            #pragma unroll
            for (int sb = 0; sb < 2; ++sb)
                #pragma unroll
                for (int r = 0; r < 8; ++r) {
                    float v = s[sb][r] * scale;
                    if (need_mask) {
                        const int col = kbase + sb * 16 + l16;
                        const int row = qw + r + half * 8;
                        if (col > row) v = -1e30f;
                    }
                    s[sb][r] = v;
                }

            // ---- online softmax: row max/sum via 16-lane butterflies ----
            float alpha[8];
            #pragma unroll
            for (int r = 0; r < 8; ++r) {
                float rm = fmaxf(s[0][r], s[1][r]);
                #pragma unroll
                for (int mm = 8; mm >= 1; mm >>= 1)
                    rm = fmaxf(rm, __shfl_xor(rm, mm, 32));
                const float m_new = fmaxf(m_i[r], rm);
                const float p0 = __expf(s[0][r] - m_new);
                const float p1 = __expf(s[1][r] - m_new);
                s[0][r] = p0; s[1][r] = p1;
                float rs = p0 + p1;
                #pragma unroll
                for (int mm = 8; mm >= 1; mm >>= 1)
                    rs += __shfl_xor(rs, mm, 32);
                alpha[r] = __expf(m_i[r] - m_new);
                l_i[r]   = l_i[r] * alpha[r] + rs;
                m_i[r]   = m_new;
            }
            #pragma unroll
            for (int c = 0; c < 8; ++c)
                #pragma unroll
                for (int r = 0; r < 8; ++r) o_acc[c][r] *= alpha[r];

            // ---- P: C-layout -> A-layout via per-wave LDS (DS ops in-order per wave) ----
            #pragma unroll
            for (int sb = 0; sb < 2; ++sb)
                #pragma unroll
                for (int r = 0; r < 8; ++r)
                    PwW[(r + half * 8) * P_STRIDE + sb * 16 + l16] = (__bf16)s[sb][r];

            v16bf pa;
            {
                const __bf16* pp = &PwW[l16 * P_STRIDE + half * 8];
                const bf16x8 lo = *(const bf16x8*)(pp);        // K = kb .. kb+7
                const bf16x8 hi = *(const bf16x8*)(pp + 16);   // K = kb+16 .. kb+23
                #pragma unroll
                for (int e = 0; e < 8; ++e) { pa[e] = lo[e]; pa[8 + e] = hi[e]; }
            }

            // ---- O += P V : preload all 8 V B-operands, then 8 independent WMMAs ----
            v16bf vbm[8];
            #pragma unroll
            for (int c = 0; c < 8; ++c) {
                const __bf16* vp = &Vt[cur][(c * 16 + l16) * VT_STRIDE + half * 8];
                const bf16x8 lo = *(const bf16x8*)(vp);
                const bf16x8 hi = *(const bf16x8*)(vp + 16);
                #pragma unroll
                for (int e = 0; e < 8; ++e) { vbm[c][e] = lo[e]; vbm[c][8 + e] = hi[e]; }
            }
            #pragma unroll
            for (int c = 0; c < 8; ++c)
                o_acc[c] = __builtin_amdgcn_wmma_f32_16x16x32_bf16(
                               false, pa, false, vbm[c], (short)0, o_acc[c], false, false);
        }

        __syncthreads();   // everyone done reading buffer `nxt` (from iteration kt-1)

        // ---- convert + stage prefetched tile into buffer `nxt` ----
        if (have_next) {
            #pragma unroll
            for (int j = 0; j < 8; ++j) {
                const int f   = stg_f + j * 512;
                const int key = f >> 7;
                const int d   = f & 127;
                bf16x4 k4;
                k4[0] = (__bf16)kpre[j][0]; k4[1] = (__bf16)kpre[j][1];
                k4[2] = (__bf16)kpre[j][2]; k4[3] = (__bf16)kpre[j][3];
                *(bf16x4*)(&Klds[nxt][key * K_STRIDE + d]) = k4;
                Vt[nxt][(d + 0) * VT_STRIDE + key] = (__bf16)vpre[j][0];
                Vt[nxt][(d + 1) * VT_STRIDE + key] = (__bf16)vpre[j][1];
                Vt[nxt][(d + 2) * VT_STRIDE + key] = (__bf16)vpre[j][2];
                Vt[nxt][(d + 3) * VT_STRIDE + key] = (__bf16)vpre[j][3];
            }
        }
        __syncthreads();   // staging visible before next iteration's compute
    }

    // ---- epilogue: O / l ----
    #pragma unroll
    for (int r = 0; r < 8; ++r) {
        const float inv_l = 1.0f / l_i[r];
        float* orow = Ob + (size_t)(qw + r + half * 8) * D_DIM;
        #pragma unroll
        for (int c = 0; c < 8; ++c)
            orow[c * 16 + l16] = o_acc[c][r] * inv_l;
    }
}

extern "C" void kernel_launch(void* const* d_in, const int* in_sizes, int n_in,
                              void* d_out, int out_size, void* d_ws, size_t ws_size,
                              hipStream_t stream) {
    const float* q = (const float*)d_in[0];
    const float* k = (const float*)d_in[1];
    const float* v = (const float*)d_in[2];
    // d_in[3] = causal mask; applied analytically in-kernel.
    float* out = (float*)d_out;
    dim3 grid(S_LEN / Q_TILE, BH_DIM);
    fa_fwd_bf16wmma<<<grid, 128, 0, stream>>>(q, k, v, out);
}
